// DifferentiableRandomForestLayer_89180700934671
// MI455X (gfx1250) — compile-verified, matrix operand order checked by reading, not run on hardware
//
#include <hip/hip_runtime.h>

// ---------------------------------------------------------------------------
// DifferentiableRandomForestLayer for MI455X (gfx1250, wave32, WMMA)
//   Stage 1: d = sigmoid(X @ W)  -> fused routing product  (kernel 1)
//   Stage 2: out = clip(route @ P / 10)                    (kernel 2)
// GEMMs on v_wmma_f32_16x16x32_bf16, f32 accumulation.
// LDS staging via GLOBAL_LOAD_ASYNC_TO_LDS_B128 (ASYNCcnt), double-buffered.
// ---------------------------------------------------------------------------

typedef __bf16 bf16;
typedef __attribute__((ext_vector_type(16))) __bf16        v16bf;
typedef __attribute__((ext_vector_type(8)))  float         v8f;
typedef __attribute__((ext_vector_type(4)))  unsigned int  u32x4;
typedef int v4i_g __attribute__((vector_size(16)));   // GCC-style vec: matches builtin proto

#define LDS_ROW 40   // bf16/row: 32 data + 8 pad = 80B -> stride 20 banks, conflict-free

union Frag { u32x4 q[2]; v16bf v; };

// Sizes
#define BATCH   4096
#define FEAT    1024
#define NTREES  10
#define NLEAFS  256
#define KTOT2   (NTREES * NLEAFS)   // 2560
#define NCLS    1000
#define NCLSPAD 1024

// Workspace layout (bytes)
#define XB_OFF   0u
#define WBT_OFF  (8u  * 1024u * 1024u)                   // Xb : 8 MB
#define PT_OFF   (WBT_OFF + 2560u * 1024u * 2u)          // WbT: 5 MB
#define RT_OFF   (PT_OFF  + 1024u * 2560u * 2u)          // PT : 5 MB
// Route: 4096*2560*2 = 20 MB -> total ~38 MB

// LDS layout (double-buffered)
#define AS_BYTES (64  * LDS_ROW * 2)    // 5120
#define BS_BYTES (256 * LDS_ROW * 2)    // 20480
#define BUF_BYTES (2 * (AS_BYTES + BS_BYTES))  // 51200

// ---------------------------------------------------------------------------
// Async global->LDS 16B copy (per lane), with sync fallback.
// ---------------------------------------------------------------------------
__device__ __forceinline__ void cp_b128(const bf16* g, bf16* l)
{
#if __has_builtin(__builtin_amdgcn_global_load_async_to_lds_b128)
    __builtin_amdgcn_global_load_async_to_lds_b128(
        (__attribute__((address_space(1))) v4i_g*)g,
        (__attribute__((address_space(3))) v4i_g*)l, 0, 0);
#else
    *(u32x4*)l = *(const u32x4*)g;
#endif
}

__device__ __forceinline__ void wait_async_all()
{
#if __has_builtin(__builtin_amdgcn_global_load_async_to_lds_b128)
#  if __has_builtin(__builtin_amdgcn_s_wait_asynccnt)
    __builtin_amdgcn_s_wait_asynccnt(0);
#  else
    asm volatile("s_wait_asynccnt 0x0" ::: "memory");
#  endif
#endif
}

// ---------------------------------------------------------------------------
// WMMA core: 64(M) x 256(N) f32 tile, BK = 32, double-buffered LDS.
// A row-major [M x K]; B stored K-contiguous per output column (B^T [N x K]).
// 8 waves: wave = (nHalf<<2)|mSub; each wave owns 8 accumulators of 16x16.
// ---------------------------------------------------------------------------
__device__ __forceinline__ void wmma_tile_64x256(
    const bf16* __restrict__ Aglob, int lda,
    const bf16* __restrict__ Bglob, int ldb,
    int K, char* smem, v8f acc[8])
{
    bf16* AsBuf[2] = { (bf16*)smem,                 (bf16*)(smem + AS_BYTES) };
    bf16* BsBuf[2] = { (bf16*)(smem + 2*AS_BYTES),  (bf16*)(smem + 2*AS_BYTES + BS_BYTES) };

    const int tid   = threadIdx.x;
    const int lane  = tid & 31;
    const int wave  = tid >> 5;
    const int half  = lane >> 4;     // K half-split per ISA 16-bit A/B layout
    const int l16   = lane & 15;
    const int mSub  = wave & 3;
    const int nHalf = wave >> 2;

    const int ar_row = tid >> 2;          // 0..63
    const int ar_col = (tid & 3) << 3;    // 0,8,16,24

    // stage tile at k0 into buffer b (5 async 16B copies per thread)
    auto stage = [&](int k0, int b) {
        cp_b128(Aglob + (size_t)ar_row * lda + k0 + ar_col,
                AsBuf[b] + ar_row * LDS_ROW + ar_col);
        const bf16* gb = Bglob + (size_t)tid * ldb + k0;
        bf16*       lb = BsBuf[b] + tid * LDS_ROW;
        cp_b128(gb,      lb);
        cp_b128(gb + 8,  lb + 8);
        cp_b128(gb + 16, lb + 16);
        cp_b128(gb + 24, lb + 24);
    };

    stage(0, 0);

    int cur = 0;
    for (int k0 = 0; k0 < K; k0 += 32, cur ^= 1) {
        wait_async_all();      // own fills of buffer `cur` have landed
        __syncthreads();       // all waves filled `cur`; all prior reads of `cur^1` done
        if (k0 + 32 < K) stage(k0 + 32, cur ^ 1);   // overlaps with compute below

        // A fragment: lane (half,l16) needs K = half*8+{0..7} and 16+half*8+{0..7}
        Frag a;
        const u32x4* ar = (const u32x4*)(AsBuf[cur] + (mSub * 16 + l16) * LDS_ROW);
        a.q[0] = ar[half];
        a.q[1] = ar[2 + half];

#pragma unroll
        for (int s = 0; s < 8; ++s) {
            Frag b;
            const u32x4* br =
                (const u32x4*)(BsBuf[cur] + ((nHalf * 8 + s) * 16 + l16) * LDS_ROW);
            b.q[0] = br[half];
            b.q[1] = br[2 + half];
            acc[s] = __builtin_amdgcn_wmma_f32_16x16x32_bf16(
                false, a.v, false, b.v, (short)0, acc[s], false, false);
        }
    }
}

// ---------------------------------------------------------------------------
// Prep kernels: bf16 conversion + layout packing
// ---------------------------------------------------------------------------
__global__ __launch_bounds__(256) void cvt_x(const float* __restrict__ x,
                                             bf16* __restrict__ xb)
{
    int i = (blockIdx.x * blockDim.x + threadIdx.x) * 4;
#pragma unroll
    for (int k = 0; k < 4; ++k) xb[i + k] = (bf16)x[i + k];
}

// WbT[(t*256+n)*1024 + f] = cnn_w[t][f][n]  (n==255 -> 0 pad)
__global__ __launch_bounds__(256) void pack_w(const float* __restrict__ w,
                                              bf16* __restrict__ wbt)
{
    int o   = blockIdx.x * blockDim.x + threadIdx.x;  // < 2560*1024
    int col = o >> 10;          // t*256+n
    int f   = o & 1023;
    int t   = col >> 8, n = col & 255;
    float v = (n < 255) ? w[((size_t)t * 1024 + f) * 255 + n] : 0.0f;
    wbt[o] = (bf16)v;
}

// PT[c*2560 + (t*256+l)] = final_probabilities[t][l][c]  (c>=1000 -> 0 pad)
__global__ __launch_bounds__(256) void pack_p(const float* __restrict__ p,
                                              bf16* __restrict__ pt)
{
    int o = blockIdx.x * blockDim.x + threadIdx.x;    // < 1024*2560
    int c = o / KTOT2;
    int k = o - c * KTOT2;
    int t = k >> 8, l = k & 255;
    float v = (c < NCLS) ? p[((size_t)t * NLEAFS + l) * NCLS + c] : 0.0f;
    pt[o] = (bf16)v;
}

// ---------------------------------------------------------------------------
// Kernel 1: per (64-row block, tree): GEMM (64x1024)x(1024x256) -> sigmoid
//           -> 8-layer routing product -> Route[b][t*256+leaf] (bf16)
// ---------------------------------------------------------------------------
__global__ __launch_bounds__(256)
void drf_gemm1_route(const bf16* __restrict__ Xb,
                     const bf16* __restrict__ WbT,
                     bf16* __restrict__ Route)
{
    __shared__ __align__(16) char smem[65536];     // staging (51200) / Dl (65536) aliased
    float* Dl = (float*)smem;                      // 64 x 256 f32, live after K loop

    const int mBase = blockIdx.x * 64;
    const int t     = blockIdx.y;

    v8f acc[8];
    v8f z = {0.f, 0.f, 0.f, 0.f, 0.f, 0.f, 0.f, 0.f};
#pragma unroll
    for (int s = 0; s < 8; ++s) acc[s] = z;

    wmma_tile_64x256(Xb  + (size_t)mBase * FEAT, FEAT,
                     WbT + (size_t)t * NLEAFS * FEAT, FEAT,
                     FEAT, smem, acc);

    __syncthreads();   // all waves done reading staging LDS before Dl overwrites it

    const int tid = threadIdx.x, lane = tid & 31, wave = tid >> 5;
    const int half = lane >> 4, l16 = lane & 15;
    const int mSub = wave & 3, nHalf = wave >> 2;

    // sigmoid epilogue -> Dl   (D layout: VGPR r -> M = r + half*8, N = l16)
#pragma unroll
    for (int s = 0; s < 8; ++s) {
        int col = (nHalf * 8 + s) * 16 + l16;
#pragma unroll
        for (int r = 0; r < 8; ++r) {
            int row = mSub * 16 + half * 8 + r;
            float x = acc[s][r];
            Dl[row * 256 + col] = 1.0f / (1.0f + __expf(-x));
        }
    }
    __syncthreads();

    // routing: 4 threads per row, 64 leaves each
    const int rloc = tid >> 2;
    bf16* out = Route + (size_t)(mBase + rloc) * KTOT2 + t * NLEAFS;
    for (int i = 0; i < 64; ++i) {
        int j = (tid & 3) * 64 + i;
        float prod = 1.0f;
#pragma unroll
        for (int l = 0; l < 8; ++l) {
            int   node = (1 << l) - 1 + (j >> (8 - l));
            float v    = Dl[rloc * 256 + node];
            prod *= (((j >> (7 - l)) & 1) == 0) ? v : (1.0f - v);
        }
        out[j] = (bf16)prod;
    }
}

// ---------------------------------------------------------------------------
// Kernel 2: GEMM (4096x2560)x(2560x1000) -> /10 -> clip -> out (f32)
// ---------------------------------------------------------------------------
__global__ __launch_bounds__(256)
void drf_gemm2_out(const bf16* __restrict__ Route,
                   const bf16* __restrict__ PT,
                   float* __restrict__ Out)
{
    __shared__ __align__(16) char smem[BUF_BYTES];

    const int mBase = blockIdx.x * 64;
    const int nBase = blockIdx.y * 256;

    v8f acc[8];
    v8f z = {0.f, 0.f, 0.f, 0.f, 0.f, 0.f, 0.f, 0.f};
#pragma unroll
    for (int s = 0; s < 8; ++s) acc[s] = z;

    wmma_tile_64x256(Route + (size_t)mBase * KTOT2, KTOT2,
                     PT    + (size_t)nBase * KTOT2, KTOT2,
                     KTOT2, smem, acc);

    const int tid = threadIdx.x, lane = tid & 31, wave = tid >> 5;
    const int half = lane >> 4, l16 = lane & 15;
    const int mSub = wave & 3, nHalf = wave >> 2;

#pragma unroll
    for (int s = 0; s < 8; ++s) {
        int c = nBase + (nHalf * 8 + s) * 16 + l16;
        if (c < NCLS) {
#pragma unroll
            for (int r = 0; r < 8; ++r) {
                int   row = mBase + mSub * 16 + half * 8 + r;
                float v   = acc[s][r] * 0.1f;
                Out[(size_t)row * NCLS + c] = fminf(fmaxf(v, 0.0f), 1.0f);
            }
        }
    }
}

// ---------------------------------------------------------------------------
extern "C" void kernel_launch(void* const* d_in, const int* in_sizes, int n_in,
                              void* d_out, int out_size, void* d_ws, size_t ws_size,
                              hipStream_t stream)
{
    const float* l_input = (const float*)d_in[0];   // (4096,1024)
    const float* cnn_w   = (const float*)d_in[1];   // (10,1024,255)
    const float* finp    = (const float*)d_in[2];   // (10,256,1000)
    float*       out     = (float*)d_out;           // (4096,1000)

    char* ws    = (char*)d_ws;
    bf16* Xb    = (bf16*)(ws + XB_OFF);
    bf16* WbT   = (bf16*)(ws + WBT_OFF);
    bf16* PT    = (bf16*)(ws + PT_OFF);
    bf16* Route = (bf16*)(ws + RT_OFF);

    cvt_x <<<(BATCH * FEAT) / (256 * 4), 256, 0, stream>>>(l_input, Xb);
    pack_w<<<(KTOT2 * FEAT)    / 256, 256, 0, stream>>>(cnn_w, WbT);
    pack_p<<<(NCLSPAD * KTOT2) / 256, 256, 0, stream>>>(finp, PT);

    drf_gemm1_route<<<dim3(BATCH / 64, NTREES), 256, 0, stream>>>(Xb, WbT, Route);
    drf_gemm2_out  <<<dim3(BATCH / 64, NCLSPAD / 256), 256, 0, stream>>>(Route, PT, out);
}
